// MoELayer_11003706212967
// MI455X (gfx1250) — matrix operand order repaired and checked by reading, not compiled, not run on hardware
//
#include <hip/hip_runtime.h>

// ---------------------------------------------------------------------------
// MoE (top-2 of 8 experts), T=16384 tokens, H=1024, DFF=4096.
//  1) convert x / fc1_w / fc2_w to bf16 once (compute-bound: 550 GFLOP routed
//     vs ~400MB traffic => bf16 WMMA path, 8x the fp32 WMMA rate)
//  2) router: top-2 + softmax, per-expert token lists
//  3) fc1/fc2: 128x128x(2x32) bf16 WMMA GEMMs, double-buffered LDS fed by
//     GLOBAL_LOAD_ASYNC_TO_LDS_B128 (ASYNCcnt) when available.
// Workspace: [counts | tok E*T | gates E*T | x_bf16 | w1_bf16 | w2_bf16 | hdn]
// Requires ws_size >= ~440 MB.
// ---------------------------------------------------------------------------

#define TOKENS   16384      // B*S
#define HDIM     1024
#define DFFDIM   4096
#define NEXP     8
#define BM       128
#define BN       128
#define LDSK     64         // K elems staged in LDS per pipeline stage
#define WMMAK    32

typedef __attribute__((ext_vector_type(16))) __bf16 v16bf;
typedef __attribute__((ext_vector_type(8)))  float  v8f;
typedef __attribute__((ext_vector_type(4)))  unsigned int uint4v;
typedef __attribute__((ext_vector_type(4)))  int v4i;

static constexpr size_t OFF_TOK   = 256;
static constexpr size_t OFF_GATES = OFF_TOK + (size_t)NEXP * TOKENS * 4;
static constexpr size_t OFF_XB    = (OFF_GATES + (size_t)NEXP * TOKENS * 4 + 255) & ~(size_t)255;
static constexpr size_t OFF_W1B   = OFF_XB  + (size_t)TOKENS * HDIM * 2;
static constexpr size_t OFF_W2B   = OFF_W1B + (size_t)NEXP * DFFDIM * HDIM * 2;
static constexpr size_t OFF_HDN   = OFF_W2B + (size_t)NEXP * HDIM * DFFDIM * 2;

// ---- CDNA5 async global->LDS copy (builtin confirmed present; param type is
// ---- a v4i pointer in global/LDS address spaces per the round-2 diagnostic) ----
#if defined(__AMDGCN__) && \
    __has_builtin(__builtin_amdgcn_global_load_async_to_lds_b128) && \
    __has_builtin(__builtin_amdgcn_s_wait_asynccnt)
typedef __attribute__((address_space(1))) v4i* glb_v4i_p;
typedef __attribute__((address_space(3))) v4i* lds_v4i_p;
#define COPY16(g, l)                                                          \
  __builtin_amdgcn_global_load_async_to_lds_b128(                             \
      (glb_v4i_p)(uintptr_t)(g), (lds_v4i_p)(l), 0, 0)
#define WAIT_ASYNC_PIPE() __builtin_amdgcn_s_wait_asynccnt(8)
#define WAIT_ASYNC_ALL()  __builtin_amdgcn_s_wait_asynccnt(0)
#else
#define COPY16(g, l) (*(uint4v*)(l) = *(const uint4v*)(g))
#define WAIT_ASYNC_PIPE()
#define WAIT_ASYNC_ALL()
#endif

__device__ __forceinline__ unsigned short f2bf(float f) {
  unsigned int u = __float_as_uint(f);
  u += 0x7FFFu + ((u >> 16) & 1u);          // round-to-nearest-even
  return (unsigned short)(u >> 16);
}
__device__ __forceinline__ unsigned int pk2bf(float a, float b) {
  return (unsigned int)f2bf(a) | ((unsigned int)f2bf(b) << 16);
}

// fp32 -> bf16 bulk conversion, 8 elems/thread
__global__ __launch_bounds__(256, 2)
void cvt_f32_bf16_kernel(const float* __restrict__ s, unsigned short* __restrict__ d, int n8) {
  int i = blockIdx.x * blockDim.x + threadIdx.x;
  const int stride = gridDim.x * blockDim.x;
  for (; i < n8; i += stride) {
    const float4* s4 = (const float4*)(s + (size_t)i * 8);
    float4 a = s4[0], b = s4[1];
    uint4v r = {pk2bf(a.x, a.y), pk2bf(a.z, a.w), pk2bf(b.x, b.y), pk2bf(b.z, b.w)};
    *(uint4v*)(d + (size_t)i * 8) = r;
  }
}

// ---------------------------------------------------------------------------
// Router: one wave per token -> 8 logits, top-2, softmax over top-2, append.
// ---------------------------------------------------------------------------
__global__ __launch_bounds__(256, 1)
void moe_router_kernel(const float* __restrict__ x,
                       const float* __restrict__ rw,
                       const float* __restrict__ rb,
                       int* __restrict__ counts,
                       int* __restrict__ tok,
                       float* __restrict__ gates) {
  const int t    = blockIdx.x * 8 + (threadIdx.x >> 5);
  const int lane = threadIdx.x & 31;
  const float* xr = x + (size_t)t * HDIM;

  float s[NEXP];
#pragma unroll
  for (int e = 0; e < NEXP; ++e) s[e] = 0.f;
  for (int k = lane; k < HDIM; k += 32) {
    const float xv = xr[k];
#pragma unroll
    for (int e = 0; e < NEXP; ++e) s[e] = fmaf(xv, rw[e * HDIM + k], s[e]);
  }
#pragma unroll
  for (int e = 0; e < NEXP; ++e) {
#pragma unroll
    for (int off = 16; off >= 1; off >>= 1) s[e] += __shfl_xor(s[e], off, 32);
  }
  if (lane == 0) {
#pragma unroll
    for (int e = 0; e < NEXP; ++e) s[e] += rb[e];
    float v0 = s[0]; int i0 = 0;
#pragma unroll
    for (int e = 1; e < NEXP; ++e) if (s[e] > v0) { v0 = s[e]; i0 = e; }
    float v1 = -3.4e38f; int i1 = 0;
#pragma unroll
    for (int e = 0; e < NEXP; ++e) if (e != i0 && s[e] > v1) { v1 = s[e]; i1 = e; }
    const float e1 = __expf(v1 - v0);
    const float inv = 1.f / (1.f + e1);
    int pos0 = atomicAdd(&counts[i0], 1);
    tok[i0 * TOKENS + pos0] = t; gates[i0 * TOKENS + pos0] = inv;
    int pos1 = atomicAdd(&counts[i1], 1);
    tok[i1 * TOKENS + pos1] = t; gates[i1 * TOKENS + pos1] = e1 * inv;
  }
}

// Fragment loaders (ISA 7.12.2 layouts; LDS tiles row-major [M|N][LDSK])
__device__ __forceinline__ v16bf load_a_frag(const unsigned short* tile, int lane, int kk) {
  const int hx = lane >> 4, mr = lane & 15;
  const unsigned short* p = tile + (size_t)mr * LDSK + kk;
  union { uint4v u[2]; v16bf v; } cv;
  cv.u[0] = *(const uint4v*)(p + hx * 8);
  cv.u[1] = *(const uint4v*)(p + 16 + hx * 8);
  return cv.v;
}
__device__ __forceinline__ v16bf load_b_frag(const unsigned short* tile, int lane, int kk) {
  const int hx = lane >> 4, nr = lane & 15;
  const unsigned short* p = tile + (size_t)nr * LDSK + kk;
  union { uint4v u[2]; v16bf v; } cv;
  cv.u[0] = *(const uint4v*)(p + hx * 8);
  cv.u[1] = *(const uint4v*)(p + 16 + hx * 8);
  return cv.v;
}

// ---------------------------------------------------------------------------
// fc1: hdn[row, n] = relu( x[tok[row]] . fc1_w[e][n] + fc1_b[e][n] )  (bf16)
// ---------------------------------------------------------------------------
__global__ __launch_bounds__(256, 1)
void moe_fc1_kernel(const unsigned short* __restrict__ xb,
                    const unsigned short* __restrict__ w1b,
                    const float* __restrict__ b1,
                    const int* __restrict__ counts,
                    const int* __restrict__ tok,
                    unsigned short* __restrict__ hdn) {
  const int e = blockIdx.z;
  const int cnt = counts[e];
  int base = 0;
#pragma unroll
  for (int i = 0; i < NEXP; ++i) if (i < e) base += counts[i];
  const int m0 = blockIdx.y * BM;
  if (m0 >= cnt) return;                        // block-uniform: EXEC stays full
  const int n0 = blockIdx.x * BN;
  const unsigned short* W = w1b + (size_t)e * DFFDIM * HDIM;

  __shared__ __align__(64) unsigned short As[2][BM][LDSK];
  __shared__ __align__(64) unsigned short Bs[2][BN][LDSK];

  const int tid = threadIdx.x, lane = tid & 31, wave = tid >> 5;
  const int wm = wave >> 2, wn = wave & 3;      // 2x4 waves: 64(M) x 32(N) each
  const int ra = tid >> 1, ca = (tid & 1) * 32; // 2 threads/row, 64B each

  int arow = m0 + ra; if (arow >= cnt) arow = cnt - 1;
  const size_t aoff = (size_t)tok[e * TOKENS + arow] * HDIM + ca;
  const size_t boff = (size_t)(n0 + ra) * HDIM + ca;

  v8f acc[4][2];
#pragma unroll
  for (int mi = 0; mi < 4; ++mi)
#pragma unroll
    for (int ni = 0; ni < 2; ++ni) acc[mi][ni] = (v8f){0,0,0,0,0,0,0,0};

  auto issue = [&](int bb, int k0) {
    const unsigned short* ga = xb + aoff + k0;
    const unsigned short* gb = W  + boff + k0;
    unsigned short* la = &As[bb][ra][ca];
    unsigned short* lb = &Bs[bb][ra][ca];
#pragma unroll
    for (int i = 0; i < 4; ++i) {
      COPY16(ga + 8 * i, la + 8 * i);
      COPY16(gb + 8 * i, lb + 8 * i);
    }
  };

  issue(0, 0);
  const int KT = HDIM / LDSK;                   // 16 stages
  for (int kt = 0; kt < KT; ++kt) {
    const int cur = kt & 1;
    if (kt + 1 < KT) { issue(cur ^ 1, (kt + 1) * LDSK); WAIT_ASYNC_PIPE(); }
    else            { WAIT_ASYNC_ALL(); }
    __syncthreads();
#pragma unroll
    for (int kk = 0; kk < LDSK; kk += WMMAK) {
      v16bf af[4], bfg[2];
#pragma unroll
      for (int mi = 0; mi < 4; ++mi) af[mi]  = load_a_frag(&As[cur][wm*64 + mi*16][0], lane, kk);
#pragma unroll
      for (int ni = 0; ni < 2; ++ni) bfg[ni] = load_b_frag(&Bs[cur][wn*32 + ni*16][0], lane, kk);
#pragma unroll
      for (int mi = 0; mi < 4; ++mi)
#pragma unroll
        for (int ni = 0; ni < 2; ++ni)
          acc[mi][ni] = __builtin_amdgcn_wmma_f32_16x16x32_bf16(
              false, af[mi], false, bfg[ni], (short)0, acc[mi][ni], false, false);
    }
    __syncthreads();
  }

  // Epilogue: bias + relu -> bf16 hdn. Fast path for full tiles.
  const int hx = lane >> 4;
  const bool full = (m0 + BM) <= cnt;
#pragma unroll
  for (int ni = 0; ni < 2; ++ni) {
    const int n = n0 + wn*32 + ni*16 + (lane & 15);
    const float bias = b1[e * DFFDIM + n];
#pragma unroll
    for (int mi = 0; mi < 4; ++mi) {
#pragma unroll
      for (int v = 0; v < 8; ++v) {
        const int grow = m0 + wm*64 + mi*16 + 8*hx + v;
        if (full || grow < cnt) {
          const float val = fmaxf(acc[mi][ni][v] + bias, 0.f);
          hdn[(size_t)(base + grow) * DFFDIM + n] = f2bf(val);
        }
      }
    }
  }
}

// ---------------------------------------------------------------------------
// fc2: out[tok[row]] += gate[row] * ( hdn[row] . fc2_w[e][n] + fc2_b[e][n] )
// ---------------------------------------------------------------------------
__global__ __launch_bounds__(256, 1)
void moe_fc2_kernel(const unsigned short* __restrict__ hdn,
                    const unsigned short* __restrict__ w2b,
                    const float* __restrict__ b2,
                    const int* __restrict__ counts,
                    const int* __restrict__ tok,
                    const float* __restrict__ gates,
                    float* __restrict__ out) {
  const int e = blockIdx.z;
  const int cnt = counts[e];
  int base = 0;
#pragma unroll
  for (int i = 0; i < NEXP; ++i) if (i < e) base += counts[i];
  const int m0 = blockIdx.y * BM;
  if (m0 >= cnt) return;
  const int n0 = blockIdx.x * BN;
  const unsigned short* W = w2b + (size_t)e * HDIM * DFFDIM;

  __shared__ __align__(64) unsigned short As[2][BM][LDSK];
  __shared__ __align__(64) unsigned short Bs[2][BN][LDSK];

  const int tid = threadIdx.x, lane = tid & 31, wave = tid >> 5;
  const int wm = wave >> 2, wn = wave & 3;
  const int ra = tid >> 1, ca = (tid & 1) * 32;

  int arow = m0 + ra; if (arow >= cnt) arow = cnt - 1;
  const size_t aoff = (size_t)(base + arow) * DFFDIM + ca;
  const size_t boff = (size_t)(n0 + ra) * DFFDIM + ca;

  v8f acc[4][2];
#pragma unroll
  for (int mi = 0; mi < 4; ++mi)
#pragma unroll
    for (int ni = 0; ni < 2; ++ni) acc[mi][ni] = (v8f){0,0,0,0,0,0,0,0};

  auto issue = [&](int bb, int k0) {
    const unsigned short* ga = hdn + aoff + k0;
    const unsigned short* gb = W   + boff + k0;
    unsigned short* la = &As[bb][ra][ca];
    unsigned short* lb = &Bs[bb][ra][ca];
#pragma unroll
    for (int i = 0; i < 4; ++i) {
      COPY16(ga + 8 * i, la + 8 * i);
      COPY16(gb + 8 * i, lb + 8 * i);
    }
  };

  issue(0, 0);
  const int KT = DFFDIM / LDSK;                 // 64 stages
  for (int kt = 0; kt < KT; ++kt) {
    const int cur = kt & 1;
    if (kt + 1 < KT) { issue(cur ^ 1, (kt + 1) * LDSK); WAIT_ASYNC_PIPE(); }
    else            { WAIT_ASYNC_ALL(); }
    __syncthreads();
#pragma unroll
    for (int kk = 0; kk < LDSK; kk += WMMAK) {
      v16bf af[4], bfg[2];
#pragma unroll
      for (int mi = 0; mi < 4; ++mi) af[mi]  = load_a_frag(&As[cur][wm*64 + mi*16][0], lane, kk);
#pragma unroll
      for (int ni = 0; ni < 2; ++ni) bfg[ni] = load_b_frag(&Bs[cur][wn*32 + ni*16][0], lane, kk);
#pragma unroll
      for (int mi = 0; mi < 4; ++mi)
#pragma unroll
        for (int ni = 0; ni < 2; ++ni)
          acc[mi][ni] = __builtin_amdgcn_wmma_f32_16x16x32_bf16(
              false, af[mi], false, bfg[ni], (short)0, acc[mi][ni], false, false);
    }
    __syncthreads();
  }

  const int hx = lane >> 4;
  const bool full = (m0 + BM) <= cnt;
#pragma unroll
  for (int ni = 0; ni < 2; ++ni) {
    const int n = n0 + wn*32 + ni*16 + (lane & 15);
    const float bias = b2[e * HDIM + n];
#pragma unroll
    for (int mi = 0; mi < 4; ++mi) {
#pragma unroll
      for (int v = 0; v < 8; ++v) {
        const int grow = m0 + wm*64 + mi*16 + 8*hx + v;
        if (full || grow < cnt) {
          const int   t_id = tok[e * TOKENS + grow];
          const float gate = gates[e * TOKENS + grow];
          unsafeAtomicAdd(&out[(size_t)t_id * HDIM + n], gate * (acc[mi][ni][v] + bias));
        }
      }
    }
  }
}

// ---------------------------------------------------------------------------
extern "C" void kernel_launch(void* const* d_in, const int* in_sizes, int n_in,
                              void* d_out, int out_size, void* d_ws, size_t ws_size,
                              hipStream_t stream) {
  const float* x  = (const float*)d_in[0];
  const float* rw = (const float*)d_in[1];
  const float* rb = (const float*)d_in[2];
  const float* w1 = (const float*)d_in[3];
  const float* b1 = (const float*)d_in[4];
  const float* w2 = (const float*)d_in[5];
  const float* b2 = (const float*)d_in[6];
  float* out = (float*)d_out;

  char* ws = (char*)d_ws;
  int*            counts = (int*)(ws);
  int*            tok    = (int*)(ws + OFF_TOK);
  float*          gates  = (float*)(ws + OFF_GATES);
  unsigned short* xb     = (unsigned short*)(ws + OFF_XB);
  unsigned short* w1b    = (unsigned short*)(ws + OFF_W1B);
  unsigned short* w2b    = (unsigned short*)(ws + OFF_W2B);
  unsigned short* hdn    = (unsigned short*)(ws + OFF_HDN);

  (void)hipMemsetAsync(counts, 0, NEXP * sizeof(int), stream);
  (void)hipMemsetAsync(out, 0, (size_t)TOKENS * HDIM * sizeof(float), stream);

  const int n8x = TOKENS * HDIM / 8;
  const int n8w = NEXP * DFFDIM * HDIM / 8;
  cvt_f32_bf16_kernel<<<n8x / 256, 256, 0, stream>>>(x,  xb,  n8x);
  cvt_f32_bf16_kernel<<<n8w / 256, 256, 0, stream>>>(w1, w1b, n8w);
  cvt_f32_bf16_kernel<<<n8w / 256, 256, 0, stream>>>(w2, w2b, n8w);

  moe_router_kernel<<<TOKENS / 8, 256, 0, stream>>>(x, rw, rb, counts, tok, gates);

  moe_fc1_kernel<<<dim3(DFFDIM / BN, TOKENS / BM, NEXP), 256, 0, stream>>>(
      xb, w1b, b1, counts, tok, hdn);

  moe_fc2_kernel<<<dim3(HDIM / BN, TOKENS / BM, NEXP), 256, 0, stream>>>(
      hdn, w2b, b2, counts, tok, gates, out);
}
